// WorkingLinearAttention_20366734917755
// MI455X (gfx1250) — matrix operand order repaired and checked by reading, not compiled, not run on hardware
//
#include <hip/hip_runtime.h>
#include <hip/hip_bf16.h>

typedef __attribute__((ext_vector_type(16))) __bf16 bf16x16;
typedef __attribute__((ext_vector_type(8)))  float  f32x8;
typedef __attribute__((ext_vector_type(4)))  unsigned int u32x4;
typedef __attribute__((ext_vector_type(8)))  int  i32x8;
typedef __attribute__((ext_vector_type(4)))  int  i32x4;

#define BDIM  1024
#define HEADS 16
#define HDIM  64
#define BATCH 4
#define SEQ   8192
#define BT    (BATCH * SEQ)

__device__ __forceinline__ int lane_id() { return (int)(threadIdx.x & 31); }
__device__ __forceinline__ __bf16 f2bf(float f) { return (__bf16)f; }

__device__ __forceinline__ f32x8 wmma_bf16(bf16x16 a, bf16x16 b, f32x8 c) {
  return __builtin_amdgcn_wmma_f32_16x16x32_bf16(false, a, false, b, (short)0, c, false, false);
}

// ---- fast fragment loads: 2x ds_load_b128 each ------------------------------
__device__ __forceinline__ bf16x16 frag_a(const __bf16* S, int stride, int row0, int k0) {
  const int l = lane_id(), half = l >> 4, m = l & 15;
  const __bf16* p = S + (row0 + m) * stride + k0 + half * 8;
  union { bf16x16 v; uint4 q[2]; } u;
  u.q[0] = *(const uint4*)p;
  u.q[1] = *(const uint4*)(p + 16);
  return u.v;
}
__device__ __forceinline__ bf16x16 frag_b(const __bf16* St, int stride, int k0, int col0) {
  const int l = lane_id(), half = l >> 4, n = l & 15;
  const __bf16* p = St + (col0 + n) * stride + k0 + half * 8;
  union { bf16x16 v; uint4 q[2]; } u;
  u.q[0] = *(const uint4*)p;
  u.q[1] = *(const uint4*)(p + 16);
  return u.v;
}

// ---- Tensor Data Mover: 2D tile load with LDS padding (clang-23 6-arg form) -
__device__ __forceinline__ void tdm_load_2d(const void* gaddr, unsigned lds_off,
                                            unsigned tensor_d0, unsigned tensor_d1,
                                            unsigned long long d0_stride,
                                            unsigned tile_d0, unsigned tile_d1,
                                            unsigned pad_int_enc, unsigned pad_amt_enc) {
  unsigned long long ga = (unsigned long long)gaddr;
  u32x4 g0;
  g0[0] = 1u;                                   // count=1, user descriptor
  g0[1] = lds_off;                              // LDS byte address
  g0[2] = (unsigned)ga;                         // global_addr[31:0]
  g0[3] = (unsigned)((ga >> 32) & 0x1FFFFFFu) | (2u << 30);  // addr[56:32] | type=2
  i32x8 g1;
  g1[0] = (int)((1u << 16)                      // data_size = 2 bytes
              | (1u << 20)                      // pad_enable
              | (pad_int_enc << 22) | (pad_amt_enc << 25));
  g1[1] = (int)((tensor_d0 & 0xFFFFu) << 16);
  g1[2] = (int)(((tensor_d0 >> 16) & 0xFFFFu) | ((tensor_d1 & 0xFFFFu) << 16));
  g1[3] = (int)(((tensor_d1 >> 16) & 0xFFFFu) | ((tile_d0 & 0xFFFFu) << 16));
  g1[4] = (int)(tile_d1 & 0xFFFFu);
  g1[5] = (int)(unsigned)(d0_stride & 0xFFFFFFFFull);
  g1[6] = (int)(unsigned)((d0_stride >> 32) & 0xFFFFu);
  g1[7] = 0;
  i32x4 gz4 = {0, 0, 0, 0};
  i32x8 gz8 = {0, 0, 0, 0, 0, 0, 0, 0};
  __builtin_amdgcn_tensor_load_to_lds(g0, g1, gz4, gz4, gz8, 0);
}
__device__ __forceinline__ unsigned lds_addr_of(const void* p) {
  return (unsigned)(unsigned long long)p;       // flat LDS addr carries offset in [31:0]
}

// ---------------- Kernel 1: qkv = x @ w_qkv, elu+1 fused on q,k --------------
// Register-prefetch pipeline: next chunk's global loads overlap current WMMAs.
__global__ __launch_bounds__(256)
void k1_qkv_gemm_act(const float* __restrict__ x, const float* __restrict__ w,
                     __bf16* __restrict__ qw, __bf16* __restrict__ kw,
                     __bf16* __restrict__ vw) {
  __shared__ alignas(16) __bf16 As[128 * 40];   // row-major [m][k]
  __shared__ alignas(16) __bf16 Bt[128 * 40];   // transposed [n][k]
  const int tid = threadIdx.x;
  const int m0 = blockIdx.x * 128;
  const int n0 = blockIdx.y * 128;
  const int wv = tid >> 5, wm = wv & 3, wn = wv >> 2;
  const int c8 = (tid & 3) * 8, r = tid >> 2;           // A staging map
  const int bn = tid & 127, kg = (tid >> 7) * 16;       // B staging map
  f32x8 acc[2][4] = {};

  float4 pa[2][2];
  float  pb[16];
  auto loadA = [&](int kk) {
#pragma unroll
    for (int rr = 0; rr < 2; ++rr) {
      const int row = rr * 64 + r;
      pa[rr][0] = *(const float4*)(x + (size_t)(m0 + row) * BDIM + kk + c8);
      pa[rr][1] = *(const float4*)(x + (size_t)(m0 + row) * BDIM + kk + c8 + 4);
    }
  };
  auto loadB = [&](int kk) {
#pragma unroll
    for (int j = 0; j < 16; ++j)
      pb[j] = w[(size_t)(kk + kg + j) * (3 * BDIM) + n0 + bn];
  };
  auto stageLDS = [&]() {
#pragma unroll
    for (int rr = 0; rr < 2; ++rr) {
      union { __bf16 h[8]; uint4 q; } st;
#pragma unroll
      for (int j = 0; j < 4; ++j) {
        st.h[j]     = f2bf(((const float*)&pa[rr][0])[j]);
        st.h[j + 4] = f2bf(((const float*)&pa[rr][1])[j]);
      }
      *(uint4*)(As + (rr * 64 + r) * 40 + c8) = st.q;
    }
    union { __bf16 h[16]; uint4 q[2]; } sb;
#pragma unroll
    for (int j = 0; j < 16; ++j) sb.h[j] = f2bf(pb[j]);
    *(uint4*)(Bt + bn * 40 + kg)     = sb.q[0];
    *(uint4*)(Bt + bn * 40 + kg + 8) = sb.q[1];
  };

  loadA(0); loadB(0);
  for (int kc = 0; kc < 32; ++kc) {
    stageLDS();
    __syncthreads();
    if (kc + 1 < 32) { loadA((kc + 1) * 32); loadB((kc + 1) * 32); }  // async
    bf16x16 af[2], bfr[4];
#pragma unroll
    for (int i = 0; i < 2; ++i) af[i] = frag_a(As, 40, wm * 32 + i * 16, 0);
#pragma unroll
    for (int j = 0; j < 4; ++j) bfr[j] = frag_b(Bt, 40, 0, wn * 64 + j * 16);
#pragma unroll
    for (int i = 0; i < 2; ++i)
#pragma unroll
      for (int j = 0; j < 4; ++j)
        acc[i][j] = wmma_bf16(af[i], bfr[j], acc[i][j]);
    __syncthreads();
  }

  const int l = tid & 31, half = l >> 4, nl = l & 15;
  const int sec = n0 / BDIM, cbase = n0 % BDIM;
  __bf16* dst = (sec == 0) ? qw : (sec == 1) ? kw : vw;
#pragma unroll
  for (int i = 0; i < 2; ++i)
#pragma unroll
    for (int j = 0; j < 4; ++j)
#pragma unroll
      for (int v = 0; v < 8; ++v) {
        const int row = m0 + wm * 32 + i * 16 + half * 8 + v;
        const int col = cbase + wn * 64 + j * 16 + nl;
        float val = acc[i][j][v];
        if (sec < 2) val = (val > 0.f) ? (val + 1.f) : __expf(val);
        dst[(size_t)row * BDIM + col] = f2bf(val);
      }
}

// ---------------- Kernel 2: kv^T per (b,h), ksum = sum_n k (pipelined) -------
__global__ __launch_bounds__(256)
void k2_kv_reduce(const __bf16* __restrict__ kw, const __bf16* __restrict__ vw,
                  __bf16* __restrict__ kvT, float* __restrict__ ksum) {
  __shared__ alignas(16) __bf16 Kt[64 * 136];   // [d][n]
  __shared__ alignas(16) __bf16 Vt[64 * 136];   // [e][n]
  __shared__ float red[4 * 64];
  const int tid = threadIdx.x;
  const int bh = blockIdx.x, b = bh >> 4, h = bh & 15;
  const __bf16* kp = kw + (size_t)b * SEQ * BDIM + h * HDIM;
  const __bf16* vp = vw + (size_t)b * SEQ * BDIM + h * HDIM;
  const int wv = tid >> 5;
  const int t0 = wv * 2, t1 = wv * 2 + 1;
  f32x8 acc0 = {}, acc1 = {};
  const int d = tid & 63, ng = (tid >> 6) * 32;
  float ks = 0.f;

  union Pack { __bf16 h[32]; uint4 q[4]; };
  Pack sk, sv;
  auto loadChunk = [&](int nt) {
#pragma unroll
    for (int j = 0; j < 32; ++j) {
      sk.h[j] = kp[(size_t)(nt + ng + j) * BDIM + d];
      sv.h[j] = vp[(size_t)(nt + ng + j) * BDIM + d];
    }
  };

  loadChunk(0);
  for (int nt = 0; nt < SEQ; nt += 128) {
#pragma unroll
    for (int j = 0; j < 4; ++j) {
      *(uint4*)(Kt + d * 136 + ng + j * 8) = sk.q[j];
      *(uint4*)(Vt + d * 136 + ng + j * 8) = sv.q[j];
    }
#pragma unroll
    for (int j = 0; j < 32; ++j) ks += (float)sk.h[j];  // ksum from regs
    __syncthreads();
    if (nt + 128 < SEQ) loadChunk(nt + 128);            // async prefetch
#pragma unroll
    for (int kc = 0; kc < 4; ++kc) {
      bf16x16 a0 = frag_a(Kt, 136, (t0 >> 2) * 16, kc * 32);
      bf16x16 b0 = frag_b(Vt, 136, kc * 32, (t0 & 3) * 16);
      acc0 = wmma_bf16(a0, b0, acc0);
      bf16x16 a1 = frag_a(Kt, 136, (t1 >> 2) * 16, kc * 32);
      bf16x16 b1 = frag_b(Vt, 136, kc * 32, (t1 & 3) * 16);
      acc1 = wmma_bf16(a1, b1, acc1);
    }
    __syncthreads();
  }

  const int l = tid & 31, half = l >> 4, nl = l & 15;
#pragma unroll
  for (int v = 0; v < 8; ++v) {                 // store TRANSPOSED: kvT[e][d]
    const int r0 = (t0 >> 2) * 16 + half * 8 + v, c0 = (t0 & 3) * 16 + nl;
    const int r1 = (t1 >> 2) * 16 + half * 8 + v, c1 = (t1 & 3) * 16 + nl;
    kvT[(size_t)bh * 4096 + c0 * 64 + r0] = f2bf(acc0[v]);
    kvT[(size_t)bh * 4096 + c1 * 64 + r1] = f2bf(acc1[v]);
  }
  red[(tid >> 6) * 64 + d] = ks;
  __syncthreads();
  if (tid < 64)
    ksum[bh * 64 + tid] = red[tid] + red[64 + tid] + red[128 + tid] + red[192 + tid];
}

// ---------------- Kernel 3: out = (q @ kv) / (q . ksum + eps)  [TDM loads] ---
__global__ __launch_bounds__(256)
void k3_apply_kv(const __bf16* __restrict__ qw, const __bf16* __restrict__ kvT,
                 const float* __restrict__ ksum, __bf16* __restrict__ attn) {
  __shared__ alignas(16) __bf16 Qs[128 * 72];
  __shared__ alignas(16) __bf16 KVs[64 * 72];
  __shared__ float ksum_s[64];
  __shared__ float inv_s[128];
  const int tid = threadIdx.x;
  const int n0 = blockIdx.x * 128;
  const int bh = blockIdx.y, b = bh >> 4, h = bh & 15;
  const __bf16* qp = qw + (size_t)(b * SEQ + n0) * BDIM + h * HDIM;

  if ((tid >> 5) == 0) {                        // wave 0 drives the TDM engine
    tdm_load_2d(qp, lds_addr_of(Qs), 64, 128, BDIM, 64, 128, 4, 3);
    tdm_load_2d(kvT + (size_t)bh * 4096, lds_addr_of(KVs), 64, 64, 64, 64, 64, 4, 3);
    __builtin_amdgcn_s_wait_tensorcnt(0);
  }
  if (tid < 64) ksum_s[tid] = ksum[bh * 64 + tid];
  __syncthreads();
  if (tid < 128) {
    float dot = 0.f;
#pragma unroll
    for (int dd = 0; dd < 64; ++dd) dot += (float)Qs[tid * 72 + dd] * ksum_s[dd];
    inv_s[tid] = 1.f / (dot + 1e-6f);
  }
  __syncthreads();
  const int wv = tid >> 5;
  f32x8 acc[4] = {};
#pragma unroll
  for (int kc = 0; kc < 2; ++kc) {
    bf16x16 a = frag_a(Qs, 72, wv * 16, kc * 32);
#pragma unroll
    for (int j = 0; j < 4; ++j)
      acc[j] = wmma_bf16(a, frag_b(KVs, 72, kc * 32, j * 16), acc[j]);
  }
  const int l = tid & 31, half = l >> 4, nl = l & 15;
#pragma unroll
  for (int j = 0; j < 4; ++j)
#pragma unroll
    for (int v = 0; v < 8; ++v) {
      const int rl = wv * 16 + half * 8 + v;
      const float val = acc[j][v] * inv_s[rl];
      attn[(size_t)(b * SEQ + n0 + rl) * BDIM + h * HDIM + j * 16 + nl] = f2bf(val);
    }
}

// ---------------- Kernel 4: out = attn @ w_out + b_out -----------------------
// Double-buffered async TDM on A; register-prefetched B conversion.
__global__ __launch_bounds__(256)
void k4_proj_gemm(const __bf16* __restrict__ a, const float* __restrict__ w,
                  const float* __restrict__ bias, float* __restrict__ out) {
  __shared__ alignas(16) __bf16 As[2][128 * 40];
  __shared__ alignas(16) __bf16 Bt[128 * 40];
  const int tid = threadIdx.x;
  const int m0 = blockIdx.x * 128;
  const int n0 = blockIdx.y * 128;
  const int wv = tid >> 5, wm = wv & 3, wn = wv >> 2;
  const int bn = tid & 127, kg = (tid >> 7) * 16;
  f32x8 acc[2][4] = {};

  float pb[16];
  auto loadB = [&](int kk) {
#pragma unroll
    for (int j = 0; j < 16; ++j)
      pb[j] = w[(size_t)(kk + kg + j) * BDIM + n0 + bn];
  };
  auto issueA = [&](int buf, int kk) {          // A tile 128x32 -> As[buf], stride 40
    tdm_load_2d(a + (size_t)m0 * BDIM + kk, lds_addr_of(As[buf]),
                32, 128, BDIM, 32, 128, /*pad 16dw*/3, /*amt 4dw*/3);
  };

  if ((tid >> 5) == 0) issueA(0, 0);
  loadB(0);
  for (int kc = 0; kc < 32; ++kc) {
    {                                           // stage B (prev readers done at barrier2)
      union { __bf16 h[16]; uint4 q[2]; } sb;
#pragma unroll
      for (int j = 0; j < 16; ++j) sb.h[j] = f2bf(pb[j]);
      *(uint4*)(Bt + bn * 40 + kg)     = sb.q[0];
      *(uint4*)(Bt + bn * 40 + kg + 8) = sb.q[1];
    }
    if (kc + 1 < 32) {
      if ((tid >> 5) == 0) issueA((kc + 1) & 1, (kc + 1) * 32);  // overlap DMA
      loadB((kc + 1) * 32);
      if ((tid >> 5) == 0) __builtin_amdgcn_s_wait_tensorcnt(1); // chunk kc done
    } else {
      if ((tid >> 5) == 0) __builtin_amdgcn_s_wait_tensorcnt(0);
    }
    __syncthreads();
    const __bf16* Ab = As[kc & 1];
    bf16x16 af[2], bfr[4];
#pragma unroll
    for (int i = 0; i < 2; ++i) af[i] = frag_a(Ab, 40, wm * 32 + i * 16, 0);
#pragma unroll
    for (int j = 0; j < 4; ++j) bfr[j] = frag_b(Bt, 40, 0, wn * 64 + j * 16);
#pragma unroll
    for (int i = 0; i < 2; ++i)
#pragma unroll
      for (int j = 0; j < 4; ++j)
        acc[i][j] = wmma_bf16(af[i], bfr[j], acc[i][j]);
    __syncthreads();
  }

  const int l = tid & 31, half = l >> 4, nl = l & 15;
#pragma unroll
  for (int i = 0; i < 2; ++i)
#pragma unroll
    for (int j = 0; j < 4; ++j)
#pragma unroll
      for (int v = 0; v < 8; ++v) {
        const int row = m0 + wm * 32 + i * 16 + half * 8 + v;
        const int col = n0 + wn * 64 + j * 16 + nl;
        out[(size_t)row * BDIM + col] = acc[i][j][v] + bias[col];
      }
}

extern "C" void kernel_launch(void* const* d_in, const int* in_sizes, int n_in,
                              void* d_out, int out_size, void* d_ws, size_t ws_size,
                              hipStream_t stream) {
  const float* x     = (const float*)d_in[0];
  const float* w_qkv = (const float*)d_in[1];
  const float* w_out = (const float*)d_in[2];
  const float* b_out = (const float*)d_in[3];

  char* ws = (char*)d_ws;
  const size_t act_bytes = (size_t)BT * BDIM * sizeof(__bf16);   // 64 MB each
  __bf16* qw   = (__bf16*)(ws);
  __bf16* kw   = (__bf16*)(ws + act_bytes);
  __bf16* vw   = (__bf16*)(ws + 2 * act_bytes);
  __bf16* attn = vw;                             // reuse: v dead after k2
  __bf16* kvT  = (__bf16*)(ws + 3 * act_bytes);
  float*  ksum = (float*)(ws + 3 * act_bytes + (size_t)64 * 4096 * sizeof(__bf16));

  dim3 blk(256);
  k1_qkv_gemm_act<<<dim3(BT / 128, 24), blk, 0, stream>>>(x, w_qkv, qw, kw, vw);
  k2_kv_reduce<<<dim3(64), blk, 0, stream>>>(kw, vw, kvT, ksum);
  k3_apply_kv<<<dim3(SEQ / 128, BATCH * HEADS), blk, 0, stream>>>(qw, kvT, ksum, attn);
  k4_proj_gemm<<<dim3(BT / 128, BDIM / 128), blk, 0, stream>>>(attn, w_out, b_out,
                                                               (float*)d_out);
}
